// FEAw_32057635897473
// MI455X (gfx1250) — matrix operand order repaired and checked by reading, not compiled, not run on hardware
//
#include <hip/hip_runtime.h>
#include <hip/hip_bf16.h>
#include <math.h>

// ---------------------------------------------------------------------------
// FEAw (wavelet frequency-enhanced attention) for MI455X / gfx1250, wave32.
// Matmul -> v_wmma_f32_16x16x32_f16; A-tile staging -> TDM tensor_load_to_lds
// (padded-row descriptor); B-tile staging -> global_load_async_to_lds_b128.
// ---------------------------------------------------------------------------

typedef __attribute__((ext_vector_type(16))) _Float16 v16h;
typedef __attribute__((ext_vector_type(8)))  _Float16 v8h;
typedef __attribute__((ext_vector_type(8)))  float    v8f;
typedef __attribute__((__vector_size__(16))) int      v4i_vs;  // async-lds builtin ptr elem
typedef unsigned int __attribute__((ext_vector_type(4))) v4u32;
typedef int __attribute__((ext_vector_type(8)))  v8i32;
typedef int __attribute__((ext_vector_type(4)))  v4i32;

#define BATCH 4
#define SCALE_C (1.0f / 262144.0f)  // 1/(512*512)

#if defined(__AMDGCN__) && __has_builtin(__builtin_amdgcn_global_load_async_to_lds_b128)
#define HAS_ASYNC_LDS 1
#else
#define HAS_ASYNC_LDS 0
#endif

#if defined(__AMDGCN__) && __has_builtin(__builtin_amdgcn_tensor_load_to_lds) && \
    __has_builtin(__builtin_amdgcn_s_wait_tensorcnt)
#define HAS_TDM 1
#else
#define HAS_TDM 0
#endif

__device__ __forceinline__ void async_copy_b128(const _Float16* g, _Float16* l) {
#if HAS_ASYNC_LDS
  __builtin_amdgcn_global_load_async_to_lds_b128((v4i_vs*)g, (v4i_vs*)l, 0, 0);
#else
  *(v8h*)l = *(const v8h*)g;
#endif
}

template <int N>
__device__ __forceinline__ void wait_async_pending() {
#if defined(__AMDGCN__)
#if __has_builtin(__builtin_amdgcn_s_wait_asynccnt)
  __builtin_amdgcn_s_wait_asynccnt(N);
#else
  if constexpr (N == 0) asm volatile("s_wait_asynccnt 0x0" ::: "memory");
  else if constexpr (N == 1) asm volatile("s_wait_asynccnt 0x1" ::: "memory");
  else if constexpr (N == 2) asm volatile("s_wait_asynccnt 0x2" ::: "memory");
  else asm volatile("s_wait_asynccnt 0x3" ::: "memory");
#endif
#endif
}

template <int N>
__device__ __forceinline__ void wait_tensor_pending() {
#if HAS_TDM
  __builtin_amdgcn_s_wait_tensorcnt(N);
#endif
}

// ---------------------------------------------------------------------------
// f32 -> f16 conversion (plain and transposing: s[R][C] -> d[C][R])
// ---------------------------------------------------------------------------
__global__ void f32_to_f16_kernel(const float* __restrict__ s,
                                  _Float16* __restrict__ d, long long n) {
  long long i = (long long)blockIdx.x * blockDim.x + threadIdx.x;
  if (i < n) d[i] = (_Float16)s[i];
}

__global__ void f32_to_f16_transpose(const float* __restrict__ s,
                                     _Float16* __restrict__ d, int R, int Cc) {
  long long i = (long long)blockIdx.x * blockDim.x + threadIdx.x;
  if (i >= (long long)R * Cc) return;
  int rr = (int)(i % R);
  int cc = (int)(i / R);
  d[(long long)cc * R + rr] = (_Float16)s[(long long)rr * Cc + cc];
}

// ---------------------------------------------------------------------------
// Truncated-DFT bases.
// Forward F (64 x t): rows 0..31 = cos(2*pi*m*i/t), rows 32..63 = -sin(...)
// Inverse Binv (t x 64): cols 0..31 = (m==0?1:2)*cos, cols 32..63 = -2*sin
// (m==0 sin column forced to 0: irfft ignores Im of bin 0). Scaled by bscale.
// ---------------------------------------------------------------------------
__global__ void fill_fwd_basis(_Float16* __restrict__ F, int t) {
  int idx = blockIdx.x * blockDim.x + threadIdx.x;
  if (idx >= 64 * t) return;
  int i = idx % t, m = idx / t;
  int mm = m & 31;
  int p = (mm * i) % t;
  float ang = 6.283185307179586f * (float)p / (float)t;
  float vv = (m < 32) ? __cosf(ang) : -__sinf(ang);
  F[idx] = (_Float16)vv;
}

__global__ void fill_inv_basis(_Float16* __restrict__ Bv, int t, float bscale) {
  int idx = blockIdx.x * blockDim.x + threadIdx.x;
  if (idx >= t * 64) return;
  int m = idx & 63, i = idx >> 6;
  int mm = m & 31;
  int p = (mm * i) % t;
  float ang = 6.283185307179586f * (float)p / (float)t;
  float vv;
  if (m < 32) vv = (mm == 0 ? 1.0f : 2.0f) * __cosf(ang);
  else        vv = (mm == 0 ? 0.0f : -2.0f) * __sinf(ang);
  Bv[idx] = (_Float16)(vv * bscale);
}

// ---------------------------------------------------------------------------
// Strided-batched GEMM: C[b] (MxN f32, +bias) = A[b] (MxK f16) * B[b] (f16).
// A tile (64x32, row stride K) staged by the Tensor Data Mover: one 2-D D#
// with LDS row padding (64B data + 16B pad) issued by wave 0, TENSORcnt.
// BT=true : B stored transposed [N][K] -> contiguous async b128 (ASYNCcnt).
// BT=false: B row-major [K][N]        -> register + LDS transpose-scatter.
// Tile 64x128x32, 256 threads = 8 waves (2x4), wave sub-tile 32x32 = 2x2 WMMA.
// Double-buffered LDS, copies pipelined one stage ahead.
// Requires M%64==0, N%128==0, K%32==0.
// ---------------------------------------------------------------------------
template <bool BT>
__global__ __launch_bounds__(256) void gemm_wmma_f16(
    const _Float16* __restrict__ A, const _Float16* __restrict__ B,
    float* __restrict__ C, const float* __restrict__ bias,
    int M, int N, int K, long long sA, long long sB, long long sC) {
  A += (long long)blockIdx.z * sA;
  B += (long long)blockIdx.z * sB;
  C += (long long)blockIdx.z * sC;

  const int m0 = blockIdx.y * 64;
  const int n0 = blockIdx.x * 128;
  const int tid = threadIdx.x;
  const int lane = tid & 31;
  const int wave = tid >> 5;
  const int wm = wave >> 2;   // 0..1 : 32-row band
  const int wn = wave & 3;    // 0..3 : 32-col band
  const int g = lane >> 4;    // lane group
  const int r = lane & 15;

  __shared__ alignas(16) _Float16 As[2][64][40];    // [stage][m][k] (+8 pad)
  __shared__ alignas(16) _Float16 Bs[2][128][40];   // [stage][n][k] (+8 pad)

  auto stageA = [&](int s, int k0) {
#if HAS_TDM
    if (tid < 32) {  // one wave issues the tile DMA (EXEC ignored by TDM)
      unsigned lds_off = (unsigned)(unsigned long long)
          (__attribute__((address_space(3))) _Float16*)&As[s][0][0];
      unsigned long long ga =
          (unsigned long long)(const void*)(A + (long long)m0 * K + k0);
      v4u32 g0;
      g0[0] = 1u;                                   // count=1 (valid user D#)
      g0[1] = lds_off;                              // lds_addr
      g0[2] = (unsigned)(ga & 0xffffffffu);         // global_addr[31:0]
      g0[3] = (unsigned)((ga >> 32) & 0x01ffffffu)  // global_addr[56:32]
              | (2u << 30);                         // type=2 ("image")
      // data_size=2B (code 1), pad_enable, pad_interval=16 DW (code 3),
      // pad_amount=4 DW (code 3): 64B row + 16B pad -> As row stride 80B.
      unsigned w0 = (1u << 16) | (1u << 20) | (3u << 22) | (3u << 25);
      unsigned td0 = (unsigned)K;    // tensor_dim0 (elements per row)
      unsigned td1 = (unsigned)M;    // tensor_dim1 (rows)
      v8i32 g1;
      g1[0] = (int)w0;
      g1[1] = (int)((td0 & 0xffffu) << 16);                   // td0[15:0]
      g1[2] = (int)((td0 >> 16) | ((td1 & 0xffffu) << 16));   // td0 hi|td1 lo
      g1[3] = (int)((td1 >> 16) | (32u << 16));               // td1 hi|tile0=32
      g1[4] = (int)64u;                                       // tile1=64
      g1[5] = (int)td0;                                       // stride0 = K
      g1[6] = 0;                                              // stride0 hi
      g1[7] = 0;
      v4i32 z4 = {0, 0, 0, 0};
      v8i32 z8 = {0, 0, 0, 0, 0, 0, 0, 0};
      __builtin_amdgcn_tensor_load_to_lds(g0, g1, z4, z4, z8, 0);
    }
#else
    int row = tid >> 2, kc = (tid & 3) * 8;
    async_copy_b128(A + (long long)(m0 + row) * K + k0 + kc, &As[s][row][kc]);
#endif
  };

  auto stageB = [&](int s, int k0) {
    if constexpr (BT) {  // B^T tile: 128x32 halfs = 512 chunks of 8
#pragma unroll
      for (int c = 0; c < 2; ++c) {
        int ch = tid + c * 256;
        int row = ch >> 2, kc = (ch & 3) * 8;
        async_copy_b128(B + (long long)(n0 + row) * K + k0 + kc, &Bs[s][row][kc]);
      }
    } else {  // B tile 32x128, transpose-scatter into [n][k]
#pragma unroll
      for (int c = 0; c < 2; ++c) {
        int ch = tid + c * 256;
        int krow = ch >> 4, nc8 = (ch & 15) * 8;
        v8h tmp = *(const v8h*)(B + (long long)(k0 + krow) * N + n0 + nc8);
#pragma unroll
        for (int j = 0; j < 8; ++j) Bs[s][nc8 + j][krow] = tmp[j];
      }
    }
  };

  auto loadA = [&](int s, int mr) -> v16h {
    v8h lo = *(const v8h*)&As[s][mr + r][g * 8];
    v8h hi = *(const v8h*)&As[s][mr + r][g * 8 + 16];
    v16h a;
#pragma unroll
    for (int i = 0; i < 8; ++i) { a[i] = lo[i]; a[8 + i] = hi[i]; }
    return a;
  };
  auto loadB = [&](int s, int nc) -> v16h {
    v8h lo = *(const v8h*)&Bs[s][nc + r][g * 8];
    v8h hi = *(const v8h*)&Bs[s][nc + r][g * 8 + 16];
    v16h b;
#pragma unroll
    for (int i = 0; i < 8; ++i) { b[i] = lo[i]; b[8 + i] = hi[i]; }
    return b;
  };

  const v8f vz = {0.f, 0.f, 0.f, 0.f, 0.f, 0.f, 0.f, 0.f};
  v8f acc[2][2] = {{vz, vz}, {vz, vz}};

  // async instructions issued per stage (per wave): A uses TDM when available
#if HAS_TDM
  constexpr int NASY = BT ? 2 : 0;
#else
  constexpr int NASY = BT ? 3 : 1;
#endif

  const int nk = K >> 5;
  stageA(0, 0);
  stageB(0, 0);
  for (int kt = 0; kt < nk; ++kt) {
    const int cur = kt & 1;
    if (kt + 1 < nk) {
      stageA(cur ^ 1, (kt + 1) << 5);
      stageB(cur ^ 1, (kt + 1) << 5);
    }
#if HAS_ASYNC_LDS || HAS_TDM
    // in-order completion: keep only the newest stage's copies outstanding
    if (kt + 1 < nk) { wait_async_pending<NASY>(); wait_tensor_pending<1>(); }
    else             { wait_async_pending<0>();    wait_tensor_pending<0>(); }
#endif
    __syncthreads();

    v16h a0 = loadA(cur, wm * 32);
    v16h a1 = loadA(cur, wm * 32 + 16);
    v16h b0 = loadB(cur, wn * 32);
    v16h b1 = loadB(cur, wn * 32 + 16);

    acc[0][0] = __builtin_amdgcn_wmma_f32_16x16x32_f16(false, a0, false, b0,
                                                       (short)0, acc[0][0], false, false);
    acc[0][1] = __builtin_amdgcn_wmma_f32_16x16x32_f16(false, a0, false, b1,
                                                       (short)0, acc[0][1], false, false);
    acc[1][0] = __builtin_amdgcn_wmma_f32_16x16x32_f16(false, a1, false, b0,
                                                       (short)0, acc[1][0], false, false);
    acc[1][1] = __builtin_amdgcn_wmma_f32_16x16x32_f16(false, a1, false, b1,
                                                       (short)0, acc[1][1], false, false);
    __syncthreads();
  }

  // epilogue: C/D fragment -> VGPR e holds row g*8+e, column = lane&15
#pragma unroll
  for (int i = 0; i < 2; ++i) {
#pragma unroll
    for (int j = 0; j < 2; ++j) {
      int row0 = m0 + wm * 32 + i * 16 + g * 8;
      int col = n0 + wn * 32 + j * 16 + r;
      float bvv = bias ? bias[col] : 0.0f;
#pragma unroll
      for (int e = 0; e < 8; ++e)
        C[(long long)(row0 + e) * N + col] = acc[i][j][e] + bvv;
    }
  }
}

// ---------------------------------------------------------------------------
// Wavelet decompose: x (4, 2n, 128, 8) f32 -> d,s (4, n, 128, 8).
// ---------------------------------------------------------------------------
__global__ void decompose_kernel(const float* __restrict__ x,
                                 const float* __restrict__ ecd,
                                 const float* __restrict__ ecs,
                                 _Float16* __restrict__ d16,
                                 _Float16* __restrict__ s16,
                                 float* __restrict__ s32, int n) {
  long long idx = (long long)blockIdx.x * blockDim.x + threadIdx.x;
  long long total = (long long)BATCH * n * 128;
  if (idx >= total) return;
  int c = (int)(idx & 127);
  long long bi = idx >> 7;
  int i = (int)(bi % n);
  int b = (int)(bi / n);
  const float* pe = x + (((long long)b * 2 * n + 2 * i) * 128 + c) * 8;
  const float* po = pe + 1024;
  float xa[16];
#pragma unroll
  for (int j = 0; j < 8; ++j) { xa[j] = pe[j]; xa[8 + j] = po[j]; }
  long long ob = idx * 8;
#pragma unroll
  for (int kw = 0; kw < 8; ++kw) {
    float dv = 0.f, sv = 0.f;
#pragma unroll
    for (int j = 0; j < 16; ++j) {
      dv += xa[j] * ecd[j * 8 + kw];
      sv += xa[j] * ecs[j * 8 + kw];
    }
    d16[ob + kw] = (_Float16)dv;
    s16[ob + kw] = (_Float16)sv;
    if (s32) s32[ob + kw] = sv;
  }
}

// ---------------------------------------------------------------------------
// Complex mode attention, one block per (b, h=kw). Xq/Xk/Xv are [b][64][1024]
// (rows 0..31 Re, 32..63 Im; col = c*8+h). Writes ZT [b][1024][64] in f16
// (B^T layout for the inverse-DFT GEMM), pre-scaled by zscale = sqrt(SCALE/t).
// tanh(a+ib) = (sinh 2a + i sin 2b) / (cosh 2a + cos 2b).
// ---------------------------------------------------------------------------
__global__ __launch_bounds__(256) void mode_attn_kernel(
    const float* __restrict__ Xq, const float* __restrict__ Xk,
    const float* __restrict__ Xv, _Float16* __restrict__ ZT, float zscale) {
  int b = blockIdx.x >> 3, h = blockIdx.x & 7;
  const long long base = (long long)b * 64 * 1024;
  __shared__ float Sre[32][32];
  __shared__ float Sim[32][32];

  for (int e = threadIdx.x; e < 1024; e += 256) {
    int x = e >> 5, y = e & 31;
    float ar = 0.f, ai = 0.f;
    for (int d = 0; d < 128; ++d) {
      int col = d * 8 + h;
      float qr = Xq[base + (long long)x * 1024 + col];
      float qi = Xq[base + (long long)(32 + x) * 1024 + col];
      float kr = Xk[base + (long long)y * 1024 + col];
      float ki = Xk[base + (long long)(32 + y) * 1024 + col];
      ar += qr * kr - qi * ki;
      ai += qr * ki + qi * kr;
    }
    float ta = 2.f * ar, tb = 2.f * ai;
    float re, im;
    if (fabsf(ta) > 20.f) {
      re = (ta > 0.f) ? 1.f : -1.f; im = 0.f;
    } else {
      float den = coshf(ta) + __cosf(tb);
      float inv = 1.0f / den;
      re = sinhf(ta) * inv;
      im = __sinf(tb) * inv;
    }
    Sre[x][y] = re;
    Sim[x][y] = im;
  }
  __syncthreads();

  for (int e = threadIdx.x; e < 4096; e += 256) {
    int d = e >> 5, x = e & 31;
    int col = d * 8 + h;
    float zr = 0.f, zi = 0.f;
#pragma unroll 8
    for (int y = 0; y < 32; ++y) {
      float vr = Xv[base + (long long)y * 1024 + col];
      float vi = Xv[base + (long long)(32 + y) * 1024 + col];
      float sr = Sre[x][y], si = Sim[x][y];
      zr += sr * vr - si * vi;
      zi += sr * vi + si * vr;
    }
    ZT[base + (long long)col * 64 + x] = (_Float16)(zr * zscale);
    ZT[base + (long long)col * 64 + 32 + x] = (_Float16)(zi * zscale);
  }
}

// ---------------------------------------------------------------------------
// Reconstruct with fused adds: xa = [la+lb | ra+rb] (each (4,n,128,8) f32),
// even/odd interleave via rc_e/rc_o -> (4, 2n, 128, 8). f32 and/or f16 out.
// ---------------------------------------------------------------------------
__global__ void reconstruct_kernel(const float* __restrict__ la,
                                   const float* __restrict__ lb,
                                   const float* __restrict__ ra,
                                   const float* __restrict__ rb,
                                   const float* __restrict__ rce,
                                   const float* __restrict__ rco,
                                   float* __restrict__ out32,
                                   _Float16* __restrict__ out16, int n) {
  long long idx = (long long)blockIdx.x * blockDim.x + threadIdx.x;
  long long total = (long long)BATCH * n * 128;
  if (idx >= total) return;
  int c = (int)(idx & 127);
  long long bi = idx >> 7;
  int i = (int)(bi % n);
  int b = (int)(bi / n);
  long long src = idx * 8;
  float xa[16];
#pragma unroll
  for (int j = 0; j < 8; ++j) {
    xa[j] = la[src + j] + lb[src + j];
    xa[8 + j] = ra[src + j] + rb[src + j];
  }
  long long oe = (((long long)b * 2 * n + 2 * i) * 128 + c) * 8;
  long long oo = oe + 1024;
#pragma unroll
  for (int kw = 0; kw < 8; ++kw) {
    float ev = 0.f, ov = 0.f;
#pragma unroll
    for (int j = 0; j < 16; ++j) {
      ev += xa[j] * rce[j * 8 + kw];
      ov += xa[j] * rco[j * 8 + kw];
    }
    if (out32) { out32[oe + kw] = ev; out32[oo + kw] = ov; }
    if (out16) { out16[oe + kw] = (_Float16)ev; out16[oo + kw] = (_Float16)ov; }
  }
}

// ---------------------------------------------------------------------------
// Host-side orchestration
// ---------------------------------------------------------------------------
static inline void launch_gemm(const _Float16* A, const _Float16* B, float* C,
                               const float* bias, int M, int N, int K,
                               long long sA, long long sB, long long sC,
                               int batch, bool bt, hipStream_t s) {
  dim3 grid(N / 128, M / 64, batch);
  if (bt)
    gemm_wmma_f16<true><<<grid, 256, 0, s>>>(A, B, C, bias, M, N, K, sA, sB, sC);
  else
    gemm_wmma_f16<false><<<grid, 256, 0, s>>>(A, B, C, bias, M, N, K, sA, sB, sC);
}

extern "C" void kernel_launch(void* const* d_in, const int* in_sizes, int n_in,
                              void* d_out, int out_size, void* d_ws,
                              size_t ws_size, hipStream_t stream) {
  const float* q   = (const float*)d_in[0];
  const float* k   = (const float*)d_in[1];
  const float* v   = (const float*)d_in[2];
  const float* Wq  = (const float*)d_in[3];
  const float* bq  = (const float*)d_in[4];
  const float* Wk  = (const float*)d_in[5];
  const float* bk  = (const float*)d_in[6];
  const float* Wv  = (const float*)d_in[7];
  const float* bv  = (const float*)d_in[8];
  const float* Wo  = (const float*)d_in[9];
  const float* bo  = (const float*)d_in[10];
  const float* ecd = (const float*)d_in[11];
  const float* ecs = (const float*)d_in[12];
  const float* rce = (const float*)d_in[13];
  const float* rco = (const float*)d_in[14];
  float* out = (float*)d_out;
  char* w = (char*)d_ws;

  // ---- static workspace layout (bytes), with region reuse; peak ~231 MB ----
  const size_t SZ_X16   = 8192ULL * 512 * 2;
  const size_t SZ_W16   = 512ULL * 1024 * 2;
  const size_t SZ_P32   = 8192ULL * 1024 * 4;
  const size_t SZ_L0_16 = 4ULL * 1024 * 1024 * 2;
  const size_t SZ_L0_32 = 4ULL * 1024 * 1024 * 4;
  const size_t SZ_L1_16 = 4ULL * 512 * 1024 * 2;
  const size_t SZ_L1_32 = 4ULL * 512 * 1024 * 4;
  const size_t SZ_XF    = 4ULL * 64 * 1024 * 4;

  const size_t o_Xq16 = 0, o_Xk16 = SZ_X16, o_Xv16 = 2 * SZ_X16;
  const size_t o_WqT = 3 * SZ_X16;                       // W^T f16 (1024x512)
  const size_t o_WkT = o_WqT + SZ_W16, o_WvT = o_WkT + SZ_W16;
  const size_t o_WoT = o_WvT + SZ_W16;                   // Wo^T f16 (512x1024)
  const size_t o_F1024 = o_WoT + SZ_W16;
  const size_t o_BI1024 = o_F1024 + 64ULL * 1024 * 2;
  const size_t o_F512 = o_BI1024 + 1024ULL * 64 * 2;
  const size_t o_BI512 = o_F512 + 64ULL * 512 * 2;
  const size_t o_Q0 = o_BI512 + 512ULL * 64 * 2;
  const size_t o_K0 = o_Q0 + SZ_P32, o_V0 = o_K0 + SZ_P32;
  const size_t o_d0q = o_V0 + SZ_P32;
  const size_t o_d0k = o_d0q + SZ_L0_16, o_d0v = o_d0k + SZ_L0_16;
  const size_t o_s0q16 = o_d0v + SZ_L0_16, o_s0k16 = o_s0q16 + SZ_L0_16;
  const size_t o_s0v16 = o_s0k16 + SZ_L0_16;
  const size_t o_s0q32 = o_s0v16 + SZ_L0_16;
  const size_t o_s0k32 = o_s0q32 + SZ_L0_32, o_s0v32 = o_s0k32 + SZ_L0_32;
  // reuse region A (over Xq/Xk/Xv16, dead after projections): feaf scratch
  const size_t o_Xfq = 0, o_Xfk = SZ_XF, o_Xfv = 2 * SZ_XF, o_ZT = 3 * SZ_XF;
  // reuse region B (over Q0/K0/V0 f32, dead after level-0 decompose):
  const size_t o_A1 = o_Q0;
  const size_t o_A2 = o_A1 + SZ_L0_32;
  const size_t o_d1q = o_A2 + SZ_L0_32;
  const size_t o_d1k = o_d1q + SZ_L1_16, o_d1v = o_d1k + SZ_L1_16;
  const size_t o_s1q = o_d1v + SZ_L1_16, o_s1k = o_s1q + SZ_L1_16;
  const size_t o_s1v = o_s1k + SZ_L1_16;
  const size_t o_B1 = o_s1v + SZ_L1_16;
  const size_t o_B2 = o_B1 + SZ_L1_32;
  const size_t o_R1 = o_B2 + SZ_L1_32;
  // reuse region C (over s0*32, dead after level-1 decompose):
  const size_t o_R016 = o_s0q32;

#define H16(off) ((_Float16*)(w + (off)))
#define F32(off) ((float*)(w + (off)))

  auto cvt = [&](const float* s, _Float16* d, long long n) {
    f32_to_f16_kernel<<<dim3((unsigned)((n + 255) / 256)), 256, 0, stream>>>(s, d, n);
  };
  auto cvtT = [&](const float* s, _Float16* d, int R, int Cc) {
    long long n = (long long)R * Cc;
    f32_to_f16_transpose<<<dim3((unsigned)((n + 255) / 256)), 256, 0, stream>>>(s, d, R, Cc);
  };

  // 1) convert activations; weights stored transposed for BT-GEMM staging
  cvt(q, H16(o_Xq16), 8192LL * 512);
  cvt(k, H16(o_Xk16), 8192LL * 512);
  cvt(v, H16(o_Xv16), 8192LL * 512);
  cvtT(Wq, H16(o_WqT), 512, 1024);
  cvtT(Wk, H16(o_WkT), 512, 1024);
  cvtT(Wv, H16(o_WvT), 512, 1024);
  cvtT(Wo, H16(o_WoT), 1024, 512);

  // 2) DFT bases (sqrt(SCALE/t) split between Z and inverse basis)
  fill_fwd_basis<<<256, 256, 0, stream>>>(H16(o_F1024), 1024);
  fill_fwd_basis<<<128, 256, 0, stream>>>(H16(o_F512), 512);
  fill_inv_basis<<<256, 256, 0, stream>>>(H16(o_BI1024), 1024,
                                          sqrtf(SCALE_C / 1024.0f));
  fill_inv_basis<<<128, 256, 0, stream>>>(H16(o_BI512), 512,
                                          sqrtf(SCALE_C / 512.0f));

  // 3) projections: (8192x512)@(512x1024)+bias -> f32   (B = W^T, BT path)
  launch_gemm(H16(o_Xq16), H16(o_WqT), F32(o_Q0), bq, 8192, 1024, 512, 0, 0, 0, 1, true, stream);
  launch_gemm(H16(o_Xk16), H16(o_WkT), F32(o_K0), bk, 8192, 1024, 512, 0, 0, 0, 1, true, stream);
  launch_gemm(H16(o_Xv16), H16(o_WvT), F32(o_V0), bv, 8192, 1024, 512, 0, 0, 0, 1, true, stream);

  // 4) level-0 decompose (t=2048 -> n=1024)
  decompose_kernel<<<2048, 256, 0, stream>>>(F32(o_Q0), ecd, ecs, H16(o_d0q),
                                             H16(o_s0q16), F32(o_s0q32), 1024);
  decompose_kernel<<<2048, 256, 0, stream>>>(F32(o_K0), ecd, ecs, H16(o_d0k),
                                             H16(o_s0k16), F32(o_s0k32), 1024);
  decompose_kernel<<<2048, 256, 0, stream>>>(F32(o_V0), ecd, ecs, H16(o_d0v),
                                             H16(o_s0v16), F32(o_s0v32), 1024);

  // feaf_lowpass: 3 forward-DFT GEMMs + mode attention + inverse GEMM
  auto feaf = [&](const _Float16* xq, const _Float16* xk, const _Float16* xv,
                  int t, size_t oF, size_t oBI, float* dst) {
    long long sB = (long long)t * 1024;
    launch_gemm(H16(oF), xq, F32(o_Xfq), nullptr, 64, 1024, t, 0, sB, 65536, BATCH, false, stream);
    launch_gemm(H16(oF), xk, F32(o_Xfk), nullptr, 64, 1024, t, 0, sB, 65536, BATCH, false, stream);
    launch_gemm(H16(oF), xv, F32(o_Xfv), nullptr, 64, 1024, t, 0, sB, 65536, BATCH, false, stream);
    mode_attn_kernel<<<32, 256, 0, stream>>>(F32(o_Xfq), F32(o_Xfk), F32(o_Xfv),
                                             H16(o_ZT), sqrtf(SCALE_C / (float)t));
    launch_gemm(H16(oBI), H16(o_ZT), dst, nullptr, t, 1024, 64, 0, 65536,
                (long long)t * 1024, BATCH, true, stream);
  };

  // 5/6) level-0 feafs: A1 = F(d0), A2 = F(s0)  (t=1024)
  feaf(H16(o_d0q), H16(o_d0k), H16(o_d0v), 1024, o_F1024, o_BI1024, F32(o_A1));
  feaf(H16(o_s0q16), H16(o_s0k16), H16(o_s0v16), 1024, o_F1024, o_BI1024, F32(o_A2));

  // 7) level-1 decompose of s-parts (n=1024 -> 512)
  decompose_kernel<<<1024, 256, 0, stream>>>(F32(o_s0q32), ecd, ecs, H16(o_d1q),
                                             H16(o_s1q), nullptr, 512);
  decompose_kernel<<<1024, 256, 0, stream>>>(F32(o_s0k32), ecd, ecs, H16(o_d1k),
                                             H16(o_s1k), nullptr, 512);
  decompose_kernel<<<1024, 256, 0, stream>>>(F32(o_s0v32), ecd, ecs, H16(o_d1v),
                                             H16(o_s1v), nullptr, 512);

  // 8) level-1 feafs: B1 = F(d1), B2 = F(s1)  (t=512)
  feaf(H16(o_d1q), H16(o_d1k), H16(o_d1v), 512, o_F512, o_BI512, F32(o_B1));
  feaf(H16(o_s1q), H16(o_s1k), H16(o_s1v), 512, o_F512, o_BI512, F32(o_B2));

  // 9) level-1 reconstruct: left = B1+B2 (out+Us1), right = B1+B2 (Ud1)
  reconstruct_kernel<<<1024, 256, 0, stream>>>(F32(o_B1), F32(o_B2), F32(o_B1),
                                               F32(o_B2), rce, rco, F32(o_R1),
                                               (_Float16*)nullptr, 512);

  // 10) level-0 reconstruct: left = R1+A1 (out+Us0), right = A1+A2 (Ud0)
  reconstruct_kernel<<<2048, 256, 0, stream>>>(F32(o_R1), F32(o_A1), F32(o_A1),
                                               F32(o_A2), rce, rco,
                                               (float*)nullptr, H16(o_R016), 1024);

  // 11) output projection: (8192x1024)@(1024x512)+bo -> d_out (f32)
  launch_gemm(H16(o_R016), H16(o_WoT), out, bo, 8192, 512, 1024, 0, 0, 0, 1, true, stream);

  (void)in_sizes; (void)n_in; (void)out_size; (void)ws_size;
#undef H16
#undef F32
}
